// Decoder_39908836114815
// MI455X (gfx1250) — matrix operand order repaired and checked by reading, not compiled, not run on hardware
//
#include <hip/hip_runtime.h>
#include <hip/hip_bf16.h>

#define T_STEPS 512
#define BATCH   512
#define IN      128
#define HID     512
#define GK      (IN + HID)    // 640: combined K for the gates GEMM
#define G4      (4 * HID)     // 2048 gate outputs

typedef __attribute__((ext_vector_type(16))) __bf16 bf16x16;
typedef __attribute__((ext_vector_type(8)))  float  f32x8;

union ABFrag { uint4 q[2]; bf16x16 v; };

__device__ __forceinline__ unsigned short f2bf(float f) {
    unsigned u = __float_as_uint(f);
    u += 0x7FFFu + ((u >> 16) & 1u);     // round-to-nearest-even
    return (unsigned short)(u >> 16);
}
__device__ __forceinline__ float sig_(float x) {
    return 1.0f / (1.0f + __expf(-x));
}
__device__ __forceinline__ float tanh_(float x) {
    return 2.0f / (1.0f + __expf(-2.0f * x)) - 1.0f;
}

// ---------------------------------------------------------------------------
// One-shot prep: bf16-convert weights & initial activations, fuse biases.
// z layout per row (BATCH rows): [0..IN) = input cols, [IN..GK) = hidden cols.
// Wc layout: [G4][GK] = [W_ih | W_hh] per gate row (bf16).
// ---------------------------------------------------------------------------
__global__ void __launch_bounds__(256)
prep_kernel(const float* __restrict__ x_last, const float* __restrict__ h0,
            const float* __restrict__ c0,
            const float* __restrict__ W_ih, const float* __restrict__ W_hh,
            const float* __restrict__ b_ih, const float* __restrict__ b_hh,
            const float* __restrict__ fc_W,
            unsigned short* __restrict__ z, unsigned short* __restrict__ Wc,
            unsigned short* __restrict__ fcWb,
            float* __restrict__ cbuf, float* __restrict__ bias)
{
    const int tid    = blockIdx.x * blockDim.x + threadIdx.x;
    const int stride = gridDim.x * blockDim.x;

    for (int i = tid; i < BATCH * IN; i += stride) {
        int b = i / IN, c = i % IN;
        z[b * GK + c] = f2bf(x_last[i]);
    }
    for (int i = tid; i < BATCH * HID; i += stride) {
        int b = i / HID, c = i % HID;
        z[b * GK + IN + c] = f2bf(h0[i]);
    }
    for (int i = tid; i < BATCH * HID; i += stride) cbuf[i] = c0[i];
    for (int i = tid; i < G4; i += stride)          bias[i] = b_ih[i] + b_hh[i];

    for (int i = tid; i < G4 * IN; i += stride) {
        int g = i / IN, k = i % IN;
        Wc[g * GK + k] = f2bf(W_ih[i]);
    }
    for (int i = tid; i < G4 * HID; i += stride) {
        int g = i / HID, k = i % HID;
        Wc[g * GK + IN + k] = f2bf(W_hh[i]);
    }
    for (int i = tid; i < IN * HID; i += stride) fcWb[i] = f2bf(fc_W[i]);
}

// ---------------------------------------------------------------------------
// Fused gates GEMM + LSTM cell. Each wave owns a 32x16 slab of the output:
// two M-tiles (batch) x one N-tile (hidden), accumulating all four gates.
// The four B (weight) fragments per K-step are reused across both M-tiles,
// halving weight-fetch bytes per FLOP: 12 b128 loads -> 8 WMMAs per K-step.
// Epilogue: sigmoid/tanh, c update in place, bf16 h into ping-pong buffer.
// ---------------------------------------------------------------------------
__global__ void __launch_bounds__(256)
lstm_gates_kernel(const unsigned short* __restrict__ z,
                  const unsigned short* __restrict__ Wc,
                  const float* __restrict__ bias,
                  float* __restrict__ cbuf,
                  unsigned short* __restrict__ zout)
{
    const int lane = threadIdx.x & 31;
    const int wave = threadIdx.x >> 5;
    const int lr   = lane & 15;       // row-in-tile (A) / col-in-tile (B,C)
    const int hi   = lane >> 4;       // lane-half select for K split
    const int m0   = blockIdx.y * 128 + (wave >> 1) * 32;  // batch slab base
    const int n0   = blockIdx.x * 32 + (wave & 1) * 16;    // hidden tile base

    ABFrag A0, A1, Bm;
    f32x8 acc[2][4];
#pragma unroll
    for (int mt = 0; mt < 2; ++mt)
#pragma unroll
        for (int g = 0; g < 4; ++g) acc[mt][g] = (f32x8){};

    const unsigned short* zrow0 = z + (size_t)(m0 + lr) * GK;
    const unsigned short* zrow1 = z + (size_t)(m0 + 16 + lr) * GK;
    const unsigned short* w[4];
#pragma unroll
    for (int g = 0; g < 4; ++g)
        w[g] = Wc + (size_t)(g * HID + n0 + lr) * GK;

    for (int k0 = 0; k0 < GK; k0 += 32) {
        // A fragments: 16-bit A layout — lane half 'hi' holds K = hi*8.. / 16+hi*8..
        A0.q[0] = *(const uint4*)(zrow0 + k0 + hi * 8);
        A0.q[1] = *(const uint4*)(zrow0 + k0 + 16 + hi * 8);
        A1.q[0] = *(const uint4*)(zrow1 + k0 + hi * 8);
        A1.q[1] = *(const uint4*)(zrow1 + k0 + 16 + hi * 8);

#pragma unroll
        for (int g = 0; g < 4; ++g) {
            const unsigned short* pb = w[g] + k0 + hi * 8;
            Bm.q[0] = *(const uint4*)(pb); Bm.q[1] = *(const uint4*)(pb + 16);
            acc[0][g] = __builtin_amdgcn_wmma_f32_16x16x32_bf16(
                false, A0.v, false, Bm.v, (short)0, acc[0][g], false, false);
            acc[1][g] = __builtin_amdgcn_wmma_f32_16x16x32_bf16(
                false, A1.v, false, Bm.v, (short)0, acc[1][g], false, false);
        }
    }

    const int   col = n0 + lr;
    const float bi  = bias[col];
    const float bf  = bias[HID + col];
    const float bg  = bias[2 * HID + col];
    const float bo  = bias[3 * HID + col];

#pragma unroll
    for (int mt = 0; mt < 2; ++mt) {
#pragma unroll
        for (int r = 0; r < 8; ++r) {
            // C/D layout: VGPR r, lane half 'hi' -> M = hi*8 + r, N = lr
            const int row = m0 + mt * 16 + hi * 8 + r;
            const float ig = sig_(acc[mt][0][r] + bi);
            const float fg = sig_(acc[mt][1][r] + bf);
            const float gg = tanh_(acc[mt][2][r] + bg);
            const float og = sig_(acc[mt][3][r] + bo);
            const size_t ci = (size_t)row * HID + col;
            const float cn = fg * cbuf[ci] + ig * gg;
            cbuf[ci] = cn;
            const float h = og * tanh_(cn);
            zout[(size_t)row * GK + IN + col] = f2bf(h);
        }
    }
}

// ---------------------------------------------------------------------------
// Output projection: out = 2*sigmoid(h·fc_W^T + fc_b) - 1.
// Writes fp32 result at reversed time index and bf16 copy into the input
// columns of the same activation buffer (feedback for the next step).
// ---------------------------------------------------------------------------
__global__ void __launch_bounds__(256)
fc_kernel(const unsigned short* __restrict__ z,     // reads hidden cols
          unsigned short* __restrict__ zx,          // writes input cols
          const unsigned short* __restrict__ fcWb,
          const float* __restrict__ fcb,
          float* __restrict__ out, int t)
{
    const int lane = threadIdx.x & 31;
    const int wave = threadIdx.x >> 5;
    const int lr   = lane & 15;
    const int hi   = lane >> 4;
    const int m0   = blockIdx.y * 64 + (wave >> 1) * 16;
    const int n0   = blockIdx.x * 32 + (wave & 1) * 16;

    ABFrag A, Bm;
    f32x8 acc = {};

    const unsigned short* zrow = z + (size_t)(m0 + lr) * GK + IN;
    const unsigned short* wrow = fcWb + (size_t)(n0 + lr) * HID;

    for (int k0 = 0; k0 < HID; k0 += 32) {
        A.q[0] = *(const uint4*)(zrow + k0 + hi * 8);
        A.q[1] = *(const uint4*)(zrow + k0 + 16 + hi * 8);
        const unsigned short* pb = wrow + k0 + hi * 8;
        Bm.q[0] = *(const uint4*)(pb); Bm.q[1] = *(const uint4*)(pb + 16);
        acc = __builtin_amdgcn_wmma_f32_16x16x32_bf16(false, A.v, false, Bm.v,
                                                      (short)0, acc, false, false);
    }

    const int   col = n0 + lr;
    const float bb  = fcb[col];
    float* orow = out + (size_t)(T_STEPS - 1 - t) * BATCH * IN;

#pragma unroll
    for (int r = 0; r < 8; ++r) {
        const int row = m0 + hi * 8 + r;
        const float v = 2.0f * sig_(acc[r] + bb) - 1.0f;
        orow[(size_t)row * IN + col] = v;
        zx[(size_t)row * GK + col]   = f2bf(v);
    }
}

// ---------------------------------------------------------------------------
extern "C" void kernel_launch(void* const* d_in, const int* in_sizes, int n_in,
                              void* d_out, int out_size, void* d_ws, size_t ws_size,
                              hipStream_t stream)
{
    (void)in_sizes; (void)n_in; (void)out_size; (void)ws_size;

    const float* x    = (const float*)d_in[0];
    // d_in[1] = enc_hiddens: unused by the recursive branch (matches reference)
    const float* h0   = (const float*)d_in[2];
    const float* c0   = (const float*)d_in[3];
    const float* W_ih = (const float*)d_in[4];
    const float* W_hh = (const float*)d_in[5];
    const float* b_ih = (const float*)d_in[6];
    const float* b_hh = (const float*)d_in[7];
    const float* fc_W = (const float*)d_in[8];
    const float* fc_b = (const float*)d_in[9];
    float* out = (float*)d_out;

    // Workspace layout (all 16B aligned; ~5.2 MB total)
    unsigned short* z0   = (unsigned short*)d_ws;                  // BATCH*GK bf16
    unsigned short* z1   = z0 + (size_t)BATCH * GK;                // BATCH*GK bf16
    unsigned short* Wc   = z1 + (size_t)BATCH * GK;                // G4*GK bf16
    unsigned short* fcWb = Wc + (size_t)G4 * GK;                   // IN*HID bf16
    float*          cbuf = (float*)(fcWb + (size_t)IN * HID);      // BATCH*HID f32
    float*          bias = cbuf + (size_t)BATCH * HID;             // G4 f32

    const float* x_last = x + (size_t)(T_STEPS - 1) * BATCH * IN;
    prep_kernel<<<256, 256, 0, stream>>>(x_last, h0, c0, W_ih, W_hh, b_ih, b_hh,
                                         fc_W, z0, Wc, fcWb, cbuf, bias);

    dim3 gGrid(HID / 32, BATCH / 128);  // 16 x 4 blocks, 8 waves, 32x16/wave
    dim3 fGrid(IN / 32, BATCH / 64);    //  4 x 8 blocks

    for (int t = 0; t < T_STEPS; ++t) {
        unsigned short* zin = (t & 1) ? z1 : z0;
        unsigned short* zot = (t & 1) ? z0 : z1;
        lstm_gates_kernel<<<gGrid, 256, 0, stream>>>(zin, Wc, bias, cbuf, zot);
        fc_kernel<<<fGrid, 256, 0, stream>>>(zot, zot, fcWb, fc_b, out, t);
    }
}
